// ACM_Framework_52012053954564
// MI455X (gfx1250) — matrix operand order repaired and checked by reading, not compiled, not run on hardware
//
#include <hip/hip_runtime.h>
#include <hip/hip_bf16.h>

// ---------------------------------------------------------------------------
// ACM graph filterbank for MI455X (gfx1250, wave32).
// Pipeline:
//   k_cvt_bf16   : x f32 -> bf16 (halves A-operand traffic, WMMA-native)
//   k_zero       : clear f32 edge accumulators + degree
//   k_gemm3_wmma : fused Z_hp / Z_lp / H_i via v_wmma_f32_16x16x32_bf16,
//                  A fragments loaded once per wave, weights LDS-swizzled
//   k_degree     : deg[dst] += 1 (global_atomic_add_f32)
//   k_aggregate  : per-edge bf16 row gather (L2-resident) + f32 atomic scatter
//   k_finalize   : means, ReLU, gate dots (shfl_xor reduce), 3x3 mix softmax,
//                  weighted combine -> out
// ---------------------------------------------------------------------------

typedef __attribute__((ext_vector_type(16))) __bf16 v16bf;
typedef __attribute__((ext_vector_type(8)))  __bf16 v8bf;
typedef __attribute__((ext_vector_type(4)))  __bf16 v4bf;
typedef __attribute__((ext_vector_type(8)))  float  v8f;
typedef __attribute__((ext_vector_type(4)))  float  v4f;

static constexpr int DCH = 128;   // DIN == DOUT == 128

// ---------- f32 -> bf16 streaming convert ----------
__global__ void k_cvt_bf16(const float* __restrict__ src, __bf16* __restrict__ dst, size_t n) {
  size_t i = ((size_t)blockIdx.x * blockDim.x + threadIdx.x) * 4;
  if (i + 3 < n) {
    v4f v = *(const v4f*)(src + i);
    v4bf o; o[0] = (__bf16)v[0]; o[1] = (__bf16)v[1]; o[2] = (__bf16)v[2]; o[3] = (__bf16)v[3];
    *(v4bf*)(dst + i) = o;
  } else {
    for (; i < n; ++i) dst[i] = (__bf16)src[i];
  }
}

// ---------- zero fill ----------
__global__ void k_zero(float* __restrict__ p, size_t n) {
  size_t i = (size_t)blockIdx.x * blockDim.x + threadIdx.x;
  const size_t st = (size_t)gridDim.x * blockDim.x;
  for (; i < n; i += st) p[i] = 0.0f;
}

// ---------- fused triple GEMM: one wave = one 16-row M tile, N=128, K=128 ----------
__global__ void __launch_bounds__(256)
k_gemm3_wmma(const __bf16* __restrict__ xb,
             const float* __restrict__ W0, const float* __restrict__ B0,
             const float* __restrict__ W1, const float* __restrict__ B1,
             const float* __restrict__ W2, const float* __restrict__ B2,
             float*  __restrict__ z_hp, __bf16* __restrict__ zb_hp,
             __bf16* __restrict__ zb_lp, float*  __restrict__ Hi,
             int nTiles)
{
  // 32 fragments (8 N-tiles x 4 k-steps) x 32 lanes x 16 bf16 = 32 KB
  __shared__ __attribute__((aligned(32))) __bf16 sW[32 * 32 * 16];

  const int tid  = threadIdx.x;
  const int wave = tid >> 5;
  const int lane = tid & 31;
  const int half = lane >> 4;   // which 16-lane half
  const int lm   = lane & 15;

  int tile = blockIdx.x * 8 + wave;
  if (tile >= nTiles) tile = nTiles - 1;   // wave-uniform clamp: EXEC stays all-ones for WMMA
  const size_t row0 = (size_t)tile * 16;

  // A fragments, ISA 7.12.2 (16-bit A 16x32): lanes 0-15 row M=lm hold K groups
  // [half*8 .. half*8+7] (VGPR0-3) and [16+half*8 ..] (VGPR4-7) of each k-step.
  v16bf a[4];
  const __bf16* ar = xb + (row0 + lm) * DCH;
  #pragma unroll
  for (int kk = 0; kk < 4; ++kk) {
    v8bf lo = *(const v8bf*)(ar + kk * 32 + half * 8);
    v8bf hi = *(const v8bf*)(ar + kk * 32 + 16 + half * 8);
    #pragma unroll
    for (int i = 0; i < 8; ++i) { a[kk][i] = lo[i]; a[kk][8 + i] = hi[i]; }
  }

  const float* Ws[3] = {W0, W1, W2};
  const float* Bs[3] = {B0, B1, B2};

  #pragma unroll
  for (int w = 0; w < 3; ++w) {
    __syncthreads();
    // Stage W[w] (row-major [K=128, N=128] f32) into LDS pre-swizzled so each
    // lane's B fragment is 32 contiguous bytes:
    //   fragment(nt,kk), lane = half*16 + (n&15), element e <-> K = kk*32 + (e>>3)*16 + half*8 + (e&7)
    for (int idx = tid; idx < DCH * DCH; idx += 256) {
      const int k = idx >> 7, n = idx & 127;
      const int kk = k >> 5, r = k & 31;
      const int hh = (r >> 3) & 1, e = (r & 7) | ((r >> 4) << 3);
      const int nt = n >> 4, c = n & 15;
      sW[(((nt * 4 + kk) * 32) + (hh * 16 + c)) * 16 + e] = (__bf16)Ws[w][idx];
    }
    __syncthreads();

    #pragma unroll
    for (int nt = 0; nt < 8; ++nt) {
      v8f acc = {};
      #pragma unroll
      for (int kk = 0; kk < 4; ++kk) {
        v16bf bf = *(const v16bf*)&sW[(((nt * 4 + kk) * 32) + lane) * 16];
        acc = __builtin_amdgcn_wmma_f32_16x16x32_bf16(false, a[kk], false, bf,
                                                      (short)0, acc, false, false);
      }
      // C/D layout: lane holds column N = lm, rows M = half*8 + r  (r = VGPR idx)
      const int n = nt * 16 + lm;
      const float bias = Bs[w][n];
      #pragma unroll
      for (int r = 0; r < 8; ++r) {
        const size_t o = (row0 + half * 8 + r) * DCH + n;
        const float v = acc[r] + bias;
        if (w == 0)      { z_hp[o] = v; zb_hp[o] = (__bf16)v; }  // f32 for (z-mean), bf16 for gather
        else if (w == 1) { zb_lp[o] = (__bf16)v; }               // only gathered per-edge
        else             { Hi[o]   = v > 0.0f ? v : 0.0f; }      // H_i = relu(xW_i + b_i)
      }
    }
  }
}

// ---------- in-degree (as float, matches reference segment_sum of ones) ----------
__global__ void k_degree(const long long* __restrict__ ei, float* __restrict__ deg, int E_) {
  const int e = blockIdx.x * blockDim.x + threadIdx.x;
  if (e < E_) atomicAdd(&deg[(int)ei[(size_t)E_ + e]], 1.0f);
}

// ---------- edge aggregation: one wave per edge-row; 4 f32 cols per lane ----------
__global__ void k_aggregate(const long long* __restrict__ ei,
                            const __bf16* __restrict__ zh, const __bf16* __restrict__ zl,
                            float* __restrict__ aggh, float* __restrict__ aggl, int E_)
{
  const int lane = threadIdx.x & 31;
  int wid = (int)((blockIdx.x * blockDim.x + threadIdx.x) >> 5);
  const int nw = (int)((gridDim.x * blockDim.x) >> 5);
  for (int e = wid; e < E_; e += nw) {
    if (e + nw < E_) __builtin_prefetch(ei + e + nw, 0, 0);      // global_prefetch_b8
    const int src = (int)ei[e];
    const int dst = (int)ei[(size_t)E_ + e];
    const size_t so = (size_t)src * DCH + lane * 4;
    const size_t dofs = (size_t)dst * DCH + lane * 4;
    v4bf h4 = *(const v4bf*)(zh + so);   // 8-byte bf16 gather, L2-resident
    v4bf l4 = *(const v4bf*)(zl + so);
    #pragma unroll
    for (int i = 0; i < 4; ++i) {
      atomicAdd(aggh + dofs + i, (float)h4[i]);                  // global_atomic_add_f32
      atomicAdd(aggl + dofs + i, (float)l4[i]);
    }
  }
}

// ---------- fused epilogue: one wave per node ----------
__global__ void k_finalize(const float* __restrict__ zhp,  const float* __restrict__ aggh,
                           const float* __restrict__ aggl, const float* __restrict__ Hi,
                           const float* __restrict__ deg,
                           const float* __restrict__ wh, const float* __restrict__ ch,
                           const float* __restrict__ wl, const float* __restrict__ cl,
                           const float* __restrict__ wi, const float* __restrict__ ci,
                           const float* __restrict__ Wm, const float* __restrict__ bm,
                           float* __restrict__ out, int N_)
{
  const int lane = threadIdx.x & 31;
  const int node = (int)((blockIdx.x * blockDim.x + threadIdx.x) >> 5);
  if (node >= N_) return;

  float d = deg[node]; if (d < 1.0f) d = 1.0f;
  const float inv = 1.0f / d;
  const size_t base = (size_t)node * DCH + lane * 4;

  v4f z  = *(const v4f*)(zhp  + base);
  v4f ah = *(const v4f*)(aggh + base);
  v4f al = *(const v4f*)(aggl + base);
  v4f hi = *(const v4f*)(Hi   + base);

  v4f Hh, Hl;
  float ph = 0.0f, pl = 0.0f, pi = 0.0f;
  #pragma unroll
  for (int i = 0; i < 4; ++i) {
    float t = z[i] - ah[i] * inv; t = t > 0.0f ? t : 0.0f; Hh[i] = t;  // HP = relu(z - mean)
    float u = al[i] * inv;        u = u > 0.0f ? u : 0.0f; Hl[i] = u;  // LP = relu(mean)
    const int c = lane * 4 + i;
    ph += t * wh[c]; pl += u * wl[c]; pi += hi[i] * wi[c];
  }
  #pragma unroll
  for (int off = 16; off > 0; off >>= 1) {        // wave32 butterfly reduce
    ph += __shfl_xor(ph, off, 32);
    pl += __shfl_xor(pl, off, 32);
    pi += __shfl_xor(pi, off, 32);
  }
  const float gh = 1.0f / (1.0f + __expf(-(ph + ch[0])));
  const float gl = 1.0f / (1.0f + __expf(-(pl + cl[0])));
  const float gi = 1.0f / (1.0f + __expf(-(pi + ci[0])));

  float m[3];
  #pragma unroll
  for (int j = 0; j < 3; ++j)
    m[j] = (gh * Wm[0 * 3 + j] + gl * Wm[1 * 3 + j] + gi * Wm[2 * 3 + j] + bm[j]) * (1.0f / 3.0f);
  const float mx = fmaxf(m[0], fmaxf(m[1], m[2]));
  const float e0 = __expf(m[0] - mx), e1 = __expf(m[1] - mx), e2 = __expf(m[2] - mx);
  const float is = 1.0f / (e0 + e1 + e2);
  const float a0 = e0 * is, a1 = e1 * is, a2 = e2 * is;

  v4f o;
  #pragma unroll
  for (int i = 0; i < 4; ++i) o[i] = a0 * Hh[i] + a1 * Hl[i] + a2 * hi[i];
  *(v4f*)(out + base) = o;
}

// ---------------------------------------------------------------------------
extern "C" void kernel_launch(void* const* d_in, const int* in_sizes, int n_in,
                              void* d_out, int out_size, void* d_ws, size_t ws_size,
                              hipStream_t stream)
{
  const float*     x   = (const float*)d_in[0];
  const long long* ei  = (const long long*)d_in[1];   // jnp.int64 edge_index [2, E]
  const float*     Whp = (const float*)d_in[2];
  const float*     bhp = (const float*)d_in[3];
  const float*     Wlp = (const float*)d_in[4];
  const float*     blp = (const float*)d_in[5];
  const float*     Wi  = (const float*)d_in[6];
  const float*     bi  = (const float*)d_in[7];
  const float*     wh  = (const float*)d_in[8];
  const float*     ch  = (const float*)d_in[9];
  const float*     wl  = (const float*)d_in[10];
  const float*     cl  = (const float*)d_in[11];
  const float*     wi2 = (const float*)d_in[12];
  const float*     ci  = (const float*)d_in[13];
  const float*     Wm  = (const float*)d_in[14];
  const float*     bm  = (const float*)d_in[15];
  float* out = (float*)d_out;

  const size_t Nn = (size_t)in_sizes[0] / DCH;   // 100000
  const size_t Ee = (size_t)in_sizes[1] / 2;     // 3200000
  const size_t ND = Nn * (size_t)DCH;

  auto up = [](size_t v) { return (v + 255) & ~(size_t)255; };
  char* p = (char*)d_ws;
  __bf16* xb    = (__bf16*)p; p += up(ND * sizeof(__bf16));   // x in bf16
  float*  z_hp  = (float*)p;  p += up(ND * sizeof(float));    // z_hp f32 (node-wise use)
  __bf16* zb_hp = (__bf16*)p; p += up(ND * sizeof(__bf16));   // z_hp bf16 (edge gather)
  __bf16* zb_lp = (__bf16*)p; p += up(ND * sizeof(__bf16));   // z_lp bf16 (edge gather)
  float*  Hi    = (float*)p;  p += up(ND * sizeof(float));    // relu(xW_i+b)
  float*  aggh  = (float*)p;  p += up(ND * sizeof(float));    // segment sums
  float*  aggl  = (float*)p;  p += up(ND * sizeof(float));
  float*  deg   = (float*)p;  p += up(Nn * sizeof(float));

  // 1) x -> bf16
  {
    const size_t n4 = (ND + 3) / 4;
    k_cvt_bf16<<<dim3((unsigned)((n4 + 255) / 256)), dim3(256), 0, stream>>>(x, xb, ND);
  }
  // 2) clear accumulators (graph-capture safe, no memset)
  k_zero<<<dim3(512), dim3(256), 0, stream>>>(aggh, ND);
  k_zero<<<dim3(512), dim3(256), 0, stream>>>(aggl, ND);
  k_zero<<<dim3(64),  dim3(256), 0, stream>>>(deg, Nn);
  // 3) fused triple GEMM via bf16 WMMA (N = 100000 is an exact multiple of 16)
  {
    const int nTiles = (int)(Nn / 16);
    const int blocks = (nTiles + 7) / 8;   // 8 waves (tiles) per 256-thread block
    k_gemm3_wmma<<<dim3(blocks), dim3(256), 0, stream>>>(xb, Whp, bhp, Wlp, blp, Wi, bi,
                                                         z_hp, zb_hp, zb_lp, Hi, nTiles);
  }
  // 4) in-degree
  k_degree<<<dim3((unsigned)((Ee + 255) / 256)), dim3(256), 0, stream>>>(ei, deg, (int)Ee);
  // 5) edge aggregation (bf16 gather, f32 L2 atomics)
  k_aggregate<<<dim3(1024), dim3(256), 0, stream>>>(ei, zb_hp, zb_lp, aggh, aggl, (int)Ee);
  // 6) fused epilogue, one wave per node
  k_finalize<<<dim3((unsigned)((Nn * 32 + 255) / 256)), dim3(256), 0, stream>>>(
      z_hp, aggh, aggl, Hi, deg, wh, ch, wl, cl, wi2, ci, Wm, bm, out, (int)Nn);
}
